// CausalSelfAttention_62070867362465
// MI455X (gfx1250) — compile-verified, matrix operand order checked by reading, not run on hardware
//
#include <hip/hip_runtime.h>

typedef __attribute__((ext_vector_type(8)))  float   v8f;
typedef __attribute__((ext_vector_type(16))) __bf16  v16bf;
typedef __attribute__((ext_vector_type(8)))  __bf16  v8bf;
typedef unsigned short u16;

constexpr int NE = 1024;   // n_embd
constexpr int NH = 16;     // heads
constexpr int HD = 64;     // head dim
constexpr int TT = 2048;   // sequence length

// ---------- helpers ----------
__device__ __forceinline__ u16 f2bf(float f) {
    union { float f; unsigned u; } v; v.f = f;
    unsigned r = v.u + 0x7FFFu + ((v.u >> 16) & 1u);   // round-to-nearest-even
    return (u16)(r >> 16);
}
__device__ __forceinline__ v8bf ldv8(const u16* p) {
    return *reinterpret_cast<const v8bf*>(p);
}
// One 16-half WMMA fragment from two contiguous 16B chunks
// (halves 0..7 = K g*8.. ; halves 8..15 = K 16+g*8.. per CDNA5 16-bit layout)
__device__ __forceinline__ v16bf frag(const u16* lo, const u16* hi) {
    v8bf a = ldv8(lo), b = ldv8(hi);
    return __builtin_shufflevector(a, b, 0,1,2,3,4,5,6,7,8,9,10,11,12,13,14,15);
}
__device__ __forceinline__ v8f wmma_bf16(v16bf a, v16bf b, v8f c) {
    return __builtin_amdgcn_wmma_f32_16x16x32_bf16(false, a, false, b, (short)0, c, false, false);
}

// ---------- precision conversion ----------
__global__ void k_f32_to_bf16(const float* __restrict__ src, u16* __restrict__ dst, int n) {
    int i = blockIdx.x * blockDim.x + threadIdx.x;
    if (i < n) dst[i] = f2bf(src[i]);
}
// src: [K,N]=[NE,NE] f32 row-major -> dst: [N,K] bf16 (B fragments then read contiguous K)
__global__ void k_transpose_bf16(const float* __restrict__ src, u16* __restrict__ dst) {
    int i = blockIdx.x * blockDim.x + threadIdx.x;
    int n = i >> 10, k = i & (NE - 1);          // NE = 1024
    dst[i] = f2bf(src[k * NE + n]);
}

// ---------- bf16 WMMA GEMM, 64x64 per wave:  out = A[M,NE] @ Bt[NE,NE]^T + bias ----------
// MODE 0: f32 out [M,NE] ; MODE 1: bf16 out [B,H,T,D] ; MODE 2: bf16 out [B,H,D,T]
template <int MODE>
__global__ void __launch_bounds__(256)
k_gemm_bf16(const u16* __restrict__ A, const u16* __restrict__ Bt,
            const float* __restrict__ bias,
            float* __restrict__ outF, u16* __restrict__ outB, int M) {
    constexpr int K = NE, N = NE;
    int lane  = threadIdx.x & 31;
    int wave  = threadIdx.x >> 5;
    int m0 = blockIdx.y * 128 + (wave >> 2) * 64;   // wave computes 64(M) x 64(N)
    int n0 = blockIdx.x * 256 + (wave & 3) * 64;
    int g   = lane >> 4;
    int c16 = lane & 15;

    const u16* arow[4];
    const u16* brow[4];
#pragma unroll
    for (int i = 0; i < 4; ++i) {
        arow[i] = A  + (size_t)(m0 + i * 16 + c16) * K + g * 8;
        brow[i] = Bt + (size_t)(n0 + i * 16 + c16) * K + g * 8;
    }

    v8f c[4][4];
#pragma unroll
    for (int mt = 0; mt < 4; ++mt)
#pragma unroll
        for (int nt = 0; nt < 4; ++nt) c[mt][nt] = (v8f){};

    for (int kb = 0; kb < K; kb += 32) {
        __builtin_prefetch(arow[0] + kb + 512, 0, 1);   // -> global_prefetch_b8
        __builtin_prefetch(brow[0] + kb + 512, 0, 1);
        v16bf a[4], b[4];
#pragma unroll
        for (int i = 0; i < 4; ++i) {
            a[i] = frag(arow[i] + kb, arow[i] + kb + 16);
            b[i] = frag(brow[i] + kb, brow[i] + kb + 16);
        }
#pragma unroll
        for (int mt = 0; mt < 4; ++mt)
#pragma unroll
            for (int nt = 0; nt < 4; ++nt)
                c[mt][nt] = wmma_bf16(a[mt], b[nt], c[mt][nt]);
    }

#pragma unroll
    for (int nt = 0; nt < 4; ++nt) {
        int nn = n0 + nt * 16 + c16;
        float bv = bias[nn];
        int h = nn >> 6, d = nn & (HD - 1);            // compile-time shifts
#pragma unroll
        for (int mt = 0; mt < 4; ++mt) {
            int rowbase = m0 + mt * 16 + g * 8;        // C-frag: VGPR r -> row g*8+r, col=lane%16
#pragma unroll
            for (int r = 0; r < 8; ++r) {
                float val = c[mt][nt][r] + bv;
                int mm = rowbase + r;
                if (MODE == 0) {
                    outF[(size_t)mm * N + nn] = val;
                } else {
                    int bb = mm >> 11, t = mm & (TT - 1);   // TT = 2048
                    if (MODE == 1) outB[(((size_t)(bb * NH + h)) * TT + t) * HD + d] = f2bf(val);
                    else           outB[(((size_t)(bb * NH + h)) * HD + d) * TT + t] = f2bf(val);
                }
            }
        }
    }
}

// ---------- flash attention: one wave per (b, h, 32-query tile) ----------
// K/V fragments for each 32-key block are reused across 2 query sub-tiles.
__global__ void __launch_bounds__(128)
k_attn(const u16* __restrict__ Q,   // [B,H,T,D] bf16
       const u16* __restrict__ Km,  // [B,H,T,D] bf16
       const u16* __restrict__ Vt,  // [B,H,D,T] bf16
       u16* __restrict__ Y) {       // [B*T, NE] bf16
    __shared__ __align__(16) u16 pbuf[4][2][16][32];   // [wave][sub][row][key]
    constexpr int nQt = TT >> 5;                        // 32 queries per wave
    int lane  = threadIdx.x & 31;
    int wslot = threadIdx.x >> 5;
    int wid   = blockIdx.x * 4 + wslot;
    int qt = wid & (nQt - 1);
    int bh = wid / nQt;
    int h  = bh & (NH - 1), b = bh >> 4;
    int g = lane >> 4, c16 = lane & 15;
    int rb = g * 8;

    // Q fragments: aq[sub][dstep]
    v16bf aq[2][2];
#pragma unroll
    for (int sub = 0; sub < 2; ++sub) {
        const u16* qb = Q + ((size_t)bh * TT + qt * 32 + sub * 16 + c16) * HD;
        aq[sub][0] = frag(qb + g * 8,      qb + 16 + g * 8);
        aq[sub][1] = frag(qb + 32 + g * 8, qb + 48 + g * 8);
    }

    float mrow[2][8], lrow[2][8];
    v8f o[2][4];
#pragma unroll
    for (int sub = 0; sub < 2; ++sub) {
#pragma unroll
        for (int r = 0; r < 8; ++r) { mrow[sub][r] = -1e30f; lrow[sub][r] = 0.f; }
#pragma unroll
        for (int nt = 0; nt < 4; ++nt) o[sub][nt] = (v8f){};
    }

    const float scale = 0.125f;                         // 1/sqrt(64)
    const u16* kbase = Km + (size_t)bh * TT * HD;
    const u16* vbase = Vt + (size_t)bh * HD * TT;

    for (int kb = 0; kb <= qt; ++kb) {                  // causal: keys up to qt*32+31
        int key0 = kb * 32;
        // ---- K fragments (shared by both query sub-tiles) ----
        v16bf bk[2][2];
#pragma unroll
        for (int nh = 0; nh < 2; ++nh) {
            const u16* kp = kbase + (size_t)(key0 + nh * 16 + c16) * HD;
            bk[nh][0] = frag(kp + g * 8,      kp + 16 + g * 8);
            bk[nh][1] = frag(kp + 32 + g * 8, kp + 48 + g * 8);
        }
        // ---- S = Q @ K^T : 8 WMMAs (2 subs x 2 key-halves x 2 d-steps) ----
        v8f s[2][2];
#pragma unroll
        for (int sub = 0; sub < 2; ++sub)
#pragma unroll
            for (int nh = 0; nh < 2; ++nh) {
                v8f acc = (v8f){};
                acc = wmma_bf16(aq[sub][0], bk[nh][0], acc);
                acc = wmma_bf16(aq[sub][1], bk[nh][1], acc);
                s[sub][nh] = acc;
            }
        // ---- online softmax per sub-tile ----
#pragma unroll
        for (int sub = 0; sub < 2; ++sub) {
            int qrow0 = qt * 32 + sub * 16 + rb;
#pragma unroll
            for (int r = 0; r < 8; ++r) {
                int qr = qrow0 + r;
                float v0 = s[sub][0][r] * scale; if (key0 + c16 > qr)      v0 = -1e30f;
                float v1 = s[sub][1][r] * scale; if (key0 + 16 + c16 > qr) v1 = -1e30f;
                float rm = fmaxf(v0, v1);
                rm = fmaxf(rm, __shfl_xor(rm, 1, 32));
                rm = fmaxf(rm, __shfl_xor(rm, 2, 32));
                rm = fmaxf(rm, __shfl_xor(rm, 4, 32));
                rm = fmaxf(rm, __shfl_xor(rm, 8, 32));  // within 16-lane half
                float mnew = fmaxf(mrow[sub][r], rm);
                float corr = __expf(mrow[sub][r] - mnew);
                float p0 = __expf(v0 - mnew);
                float p1 = __expf(v1 - mnew);
                float rs = p0 + p1;
                rs += __shfl_xor(rs, 1, 32);
                rs += __shfl_xor(rs, 2, 32);
                rs += __shfl_xor(rs, 4, 32);
                rs += __shfl_xor(rs, 8, 32);
                lrow[sub][r] = lrow[sub][r] * corr + rs;
                mrow[sub][r] = mnew;
#pragma unroll
                for (int nt = 0; nt < 4; ++nt) o[sub][nt][r] *= corr;
                pbuf[wslot][sub][rb + r][c16]      = f2bf(p0);  // C-layout -> [row][key]
                pbuf[wslot][sub][rb + r][16 + c16] = f2bf(p1);
            }
        }
        asm volatile("s_wait_dscnt 0" ::: "memory");    // per-wave LDS RAW fence

        // ---- O += P @ V : 8 WMMAs, V fragments shared across sub-tiles ----
        v16bf pA[2];
        pA[0] = frag(&pbuf[wslot][0][c16][g * 8], &pbuf[wslot][0][c16][16 + g * 8]);
        pA[1] = frag(&pbuf[wslot][1][c16][g * 8], &pbuf[wslot][1][c16][16 + g * 8]);
#pragma unroll
        for (int nt = 0; nt < 4; ++nt) {
            const u16* vp = vbase + (size_t)(nt * 16 + c16) * TT + key0;
            v16bf vb = frag(vp + g * 8, vp + 16 + g * 8);
            o[0][nt] = wmma_bf16(pA[0], vb, o[0][nt]);
            o[1][nt] = wmma_bf16(pA[1], vb, o[1][nt]);
        }
    }

    // ---- normalize + store y as [B*T, NE] bf16 ----
#pragma unroll
    for (int sub = 0; sub < 2; ++sub) {
        int mglob = b * TT + qt * 32 + sub * 16;
#pragma unroll
        for (int r = 0; r < 8; ++r) {
            float inv = 1.0f / lrow[sub][r];
            size_t row = (size_t)(mglob + rb + r) * NE + h * HD;
            Y[row +  0 + c16] = f2bf(o[sub][0][r] * inv);
            Y[row + 16 + c16] = f2bf(o[sub][1][r] * inv);
            Y[row + 32 + c16] = f2bf(o[sub][2][r] * inv);
            Y[row + 48 + c16] = f2bf(o[sub][3][r] * inv);
        }
    }
}

// ---------- host launch ----------
extern "C" void kernel_launch(void* const* d_in, const int* in_sizes, int n_in,
                              void* d_out, int out_size, void* d_ws, size_t ws_size,
                              hipStream_t stream) {
    const float* x  = (const float*)d_in[0];
    const float* Wq = (const float*)d_in[1];
    const float* bq = (const float*)d_in[2];
    const float* Wk = (const float*)d_in[3];
    const float* bk = (const float*)d_in[4];
    const float* Wv = (const float*)d_in[5];
    const float* bv = (const float*)d_in[6];
    const float* Wp = (const float*)d_in[7];
    const float* bp = (const float*)d_in[8];

    int M = in_sizes[0] / NE;     // B*T = 4096
    int B = M / TT;

    char* w = (char*)d_ws;
    u16* xb   = (u16*)w; w += (size_t)M * NE * 2;
    u16* wqt  = (u16*)w; w += (size_t)NE * NE * 2;
    u16* wkt  = (u16*)w; w += (size_t)NE * NE * 2;
    u16* wvt  = (u16*)w; w += (size_t)NE * NE * 2;
    u16* wpt  = (u16*)w; w += (size_t)NE * NE * 2;
    u16* qb   = (u16*)w; w += (size_t)M * NE * 2;
    u16* kbuf = (u16*)w; w += (size_t)M * NE * 2;
    u16* vt   = (u16*)w; w += (size_t)M * NE * 2;
    u16* yb   = xb;               // reuse: attention does not read xb

    // 1) conversions
    k_f32_to_bf16<<<(M * NE + 255) / 256, 256, 0, stream>>>(x, xb, M * NE);
    k_transpose_bf16<<<(NE * NE) / 256, 256, 0, stream>>>(Wq, wqt);
    k_transpose_bf16<<<(NE * NE) / 256, 256, 0, stream>>>(Wk, wkt);
    k_transpose_bf16<<<(NE * NE) / 256, 256, 0, stream>>>(Wv, wvt);
    k_transpose_bf16<<<(NE * NE) / 256, 256, 0, stream>>>(Wp, wpt);

    // 2) QKV projections (WMMA, 64x64 per wave; block tile 128x256)
    dim3 gg(NE / 256, M / 128), bb(256);
    k_gemm_bf16<1><<<gg, bb, 0, stream>>>(xb, wqt, bq, nullptr, qb,   M);
    k_gemm_bf16<1><<<gg, bb, 0, stream>>>(xb, wkt, bk, nullptr, kbuf, M);
    k_gemm_bf16<2><<<gg, bb, 0, stream>>>(xb, wvt, bv, nullptr, vt,   M);

    // 3) flash attention (WMMA): B*H*(T/32) waves, 4 waves/block
    int nwaves = B * NH * (TT / 32);
    k_attn<<<nwaves / 4, 128, 0, stream>>>(qb, kbuf, vt, yb);

    // 4) output projection (WMMA) -> f32 d_out
    k_gemm_bf16<0><<<gg, bb, 0, stream>>>(yb, wpt, bp, (float*)d_out, nullptr, M);

    (void)n_in; (void)out_size; (void)ws_size; (void)B;
}
